// FunnelRelMultiheadAttention_55336358643180
// MI455X (gfx1250) — matrix operand order repaired and checked by reading, not compiled, not run on hardware
//
#include <hip/hip_runtime.h>
#include <hip/hip_bf16.h>

#define BB 2
#define SS 1024
#define DD 768
#define NN 12
#define HH 64
#define TT 2048
#define NH 768
#define SCALE 0.125f
#define BIGINF 1000000.0f

typedef __attribute__((ext_vector_type(16))) _Float16 v16h;
typedef __attribute__((ext_vector_type(8)))  float    v8f;

union AF { v16h h; unsigned int u[8]; };

__device__ __forceinline__ unsigned int pack2(float x, float y) {
  union { _Float16 h[2]; unsigned int u; } t;
  t.h[0] = (_Float16)x; t.h[1] = (_Float16)y;
  return t.u;
}

// A-fragment (16x32 f16), lane row m=l%16. Contiguous runs:
//   u[0..3] = K [hi*8, hi*8+7], u[4..7] = K [16+hi*8, 16+hi*8+7]
__device__ __forceinline__ void loadA16(AF& a, const _Float16* p, int hi) {
  const _Float16* b0 = p + (hi ? 8 : 0);
  *(uint4*)&a.u[0] = *(const uint4*)(b0);
  *(uint4*)&a.u[4] = *(const uint4*)(b0 + 16);
}
// B-fragment (32x16 f16), lane col n=l%16. Contiguous runs:
//   u[0..3] = K [hi*8, hi*8+7], u[4..7] = K [16+hi*8, 16+hi*8+7]
__device__ __forceinline__ void loadB16(AF& b, const _Float16* p, int hi) {
  const _Float16* b0 = p + (hi ? 8 : 0);
  *(uint4*)&b.u[0] = *(const uint4*)(b0);
  *(uint4*)&b.u[4] = *(const uint4*)(b0 + 16);
}
// A-fragment from f32 memory: four float4 loads + cvt_pk_f16_f32 packs
__device__ __forceinline__ void loadA32(AF& a, const float* p, int hi) {
  const float* b0 = p + (hi ? 8 : 0);
  float4 f0 = *(const float4*)(b0);
  float4 f1 = *(const float4*)(b0 + 4);
  float4 f2 = *(const float4*)(b0 + 16);
  float4 f3 = *(const float4*)(b0 + 20);
  a.u[0] = pack2(f0.x, f0.y); a.u[1] = pack2(f0.z, f0.w);
  a.u[2] = pack2(f1.x, f1.y); a.u[3] = pack2(f1.z, f1.w);
  a.u[4] = pack2(f2.x, f2.y); a.u[5] = pack2(f2.z, f2.w);
  a.u[6] = pack2(f3.x, f3.y); a.u[7] = pack2(f3.z, f3.w);
}

__device__ __forceinline__ v8f wmma16(const AF& a, const AF& b, v8f c) {
  return __builtin_amdgcn_wmma_f32_16x16x32_f16(false, a.h, false, b.h,
                                                (short)0, c, false, false);
}

// ---------------- Kernel 1: transpose weights to [col][k], f32 -> f16 --------
__global__ void transpose_w(const float* __restrict__ wq, const float* __restrict__ wk,
                            const float* __restrict__ wv, const float* __restrict__ rk,
                            const float* __restrict__ wo,
                            _Float16* wqT, _Float16* wkT, _Float16* wvT,
                            _Float16* rkT, _Float16* woT) {
  int id = blockIdx.y;
  const float* src = (id == 0) ? wq : (id == 1) ? wk : (id == 2) ? wv : (id == 3) ? rk : wo;
  _Float16* dst = (id == 0) ? wqT : (id == 1) ? wkT : (id == 2) ? wvT : (id == 3) ? rkT : woT;
  int e = blockIdx.x * 256 + threadIdx.x;        // e = k*768 + c
  int k = e / DD, c = e % DD;                    // all matrices are 768x768
  dst[(size_t)c * DD + k] = (_Float16)src[e];
}

// ------------- Kernel 2: QKV + r_head projection GEMMs (WMMA) ---------------
__global__ void proj_gemm(const float* __restrict__ query, const float* __restrict__ key,
                          const float* __restrict__ value, const float* __restrict__ r,
                          const _Float16* __restrict__ wqT, const _Float16* __restrict__ wkT,
                          const _Float16* __restrict__ wvT, const _Float16* __restrict__ rkT,
                          const float* __restrict__ bk, const float* __restrict__ bv,
                          const float* __restrict__ rwb, const float* __restrict__ rrb,
                          const float* __restrict__ rsb,
                          _Float16* qw, _Float16* qr, _Float16* qs,
                          _Float16* kb, _Float16* vT, _Float16* rh) {
  int lane = threadIdx.x & 31, w = threadIdx.x >> 5;
  int it = blockIdx.x * 4 + w;        // row tile 0..127 (M=2048)
  int ct = blockIdx.y;                // col tile 0..47  (N=768)
  int id = blockIdx.z;                // 0=q 1=k 2=v 3=r
  const float* A = (id == 0) ? query : (id == 1) ? key : (id == 2) ? value : r;
  const _Float16* W = (id == 0) ? wqT : (id == 1) ? wkT : (id == 2) ? wvT : rkT;
  int colg = lane & 15, hi2 = lane >> 4;
  int row0 = it * 16;
  int ncol = ct * 16 + colg;
  const float* arow = &A[(size_t)(row0 + colg) * DD];
  const _Float16* wrow = &W[(size_t)ncol * DD];
  v8f acc = {};
  for (int kk = 0; kk < 24; ++kk) {
    int K0 = kk * 32;
    AF a, bf;
    loadA32(a, arow + K0, hi2);
    loadB16(bf, wrow + K0, hi2);
    acc = wmma16(a, bf, acc);
  }
#pragma unroll
  for (int vg = 0; vg < 8; ++vg) {
    int mrow = row0 + vg + 8 * hi2;
    int col = ct * 16 + colg;
    int nh = col >> 6, h = col & 63;
    float v = acc[vg];
    if (id == 0) {
      int b = mrow >> 10, i = mrow & (SS - 1);
      size_t o = (((size_t)(b * NN + nh)) * SS + i) * HH + h;
      qw[o] = (_Float16)((v + rwb[nh * HH + h]) * SCALE);
      qr[o] = (_Float16)((v + rrb[nh * HH + h]) * SCALE);
      qs[o] = (_Float16)((v + rsb[nh * HH + h]) * SCALE);
    } else if (id == 1) {
      int b = mrow >> 10, i = mrow & (SS - 1);
      kb[(((size_t)(b * NN + nh)) * SS + i) * HH + h] = (_Float16)(v + bk[col]);
    } else if (id == 2) {
      int b = mrow >> 10, i = mrow & (SS - 1);
      vT[(((size_t)(b * NN + nh)) * HH + h) * SS + i] = (_Float16)(v + bv[col]);
    } else {
      rh[(((size_t)nh) * TT + mrow) * HH + h] = (_Float16)v;   // mrow == t
    }
  }
}

// --------- Kernel 3: token-type biases (rank-1 per row, seg_embed dots) -----
__global__ void tt_kernel(const _Float16* __restrict__ qs, const float* __restrict__ seg,
                          float* ttS, float* ttD) {
  int idx = blockIdx.x * 256 + threadIdx.x;     // B*N*S = 24576
  if (idx >= BB * NN * SS) return;
  int i = idx & (SS - 1);
  int bn = idx >> 10;
  int n = bn % NN;
  const _Float16* q = &qs[((size_t)bn * SS + i) * HH];
  float d0 = 0.f, d1 = 0.f;
#pragma unroll 8
  for (int h = 0; h < HH; ++h) {
    float qv = (float)q[h];
    d0 += qv * seg[n * HH + h];          // seg_embed[0][n][h]
    d1 += qv * seg[NH + n * HH + h];     // seg_embed[1][n][h]
  }
  ttD[idx] = d0;
  ttS[idx] = d1;
}

// --------- Kernel 4: fused attention (content + rel-pos + tt + softmax + PV) -
__global__ void attn_kernel(const _Float16* __restrict__ qw, const _Float16* __restrict__ qr,
                            const _Float16* __restrict__ kb, const _Float16* __restrict__ vT,
                            const _Float16* __restrict__ rh,
                            const float* __restrict__ ttSame, const float* __restrict__ ttDiff,
                            const float* __restrict__ cls_mask, const int* __restrict__ token_type,
                            const int* __restrict__ amask, _Float16* av) {
  __shared__ __align__(16) float panelS[4][16 * 48];
  __shared__ __align__(16) _Float16 pstageS[4][16 * 32];
  int lane = threadIdx.x & 31, w = threadIdx.x >> 5;
  float* panel = panelS[w];
  _Float16* pstage = pstageS[w];
  int tile = blockIdx.x * 4 + w;               // bn*64 + it, 1536 total
  int it = tile & 63;
  int bn = tile >> 6;
  int n = bn % NN, b = bn / NN;
  int i0 = it * 16;
  int colg = lane & 15, hi2 = lane >> 4;

  // hoisted q fragments (A-matrices for content and pos)
  size_t qbase = ((size_t)bn * SS + i0 + colg) * HH;
  AF aw[2], ar[2];
#pragma unroll
  for (int kk = 0; kk < 2; ++kk) {
    loadA16(aw[kk], &qw[qbase + kk * 32], hi2);
    loadA16(ar[kk], &qr[qbase + kk * 32], hi2);
  }

  float ttS[8], ttD[8], m_r[8], l_r[8];
#pragma unroll
  for (int vg = 0; vg < 8; ++vg) {
    int i = i0 + vg + 8 * hi2;
    ttS[vg] = ttSame[bn * SS + i];
    ttD[vg] = ttDiff[bn * SS + i];
    m_r[vg] = -1e30f;
    l_r[vg] = 0.f;
  }
  v8f o0 = {}, o1 = {}, o2 = {}, o3 = {};
  const _Float16* kbb = &kb[(size_t)bn * SS * HH];
  const _Float16* vbb = &vT[(size_t)bn * HH * SS];
  const _Float16* rhb = &rh[(size_t)n * TT * HH];

  for (int j0 = 0; j0 < SS; j0 += 32) {
    // prefetch next iteration's K/V tiles (global_prefetch_b8; L2-resident)
    if (j0 + 32 < SS) {
      __builtin_prefetch(&kbb[(size_t)(j0 + 32 + colg + 16 * hi2) * HH], 0, 1);
      __builtin_prefetch(&vbb[(size_t)(colg + 16 * hi2) * SS + j0 + 32], 0, 1);
    }

    // ---- content scores: two 16x16 tiles over K=H=64 ----
    v8f s0 = {}, s1 = {};
#pragma unroll
    for (int kk = 0; kk < 2; ++kk) {
      AF bf0, bf1;
      loadB16(bf0, &kbb[(size_t)(j0 + colg) * HH + kk * 32], hi2);
      loadB16(bf1, &kbb[(size_t)(j0 + 16 + colg) * HH + kk * 32], hi2);
      s0 = wmma16(aw[kk], bf0, s0);
      s1 = wmma16(aw[kk], bf1, s1);
    }

    // ---- relative-position panel: 16x48 over t = S + j - i (shifted gather) -
    int tbase = SS + j0 - i0 - 16;               // always keeps t in [0,2048)
#pragma unroll
    for (int pt = 0; pt < 48; pt += 16) {
      v8f pp = {};
      int t = tbase + pt + colg;
#pragma unroll
      for (int kk = 0; kk < 2; ++kk) {
        AF bf;
        loadB16(bf, &rhb[(size_t)t * HH + kk * 32], hi2);
        pp = wmma16(ar[kk], bf, pp);
      }
#pragma unroll
      for (int vg = 0; vg < 8; ++vg)
        panel[(vg + 8 * hi2) * 48 + pt + colg] = pp[vg];
    }

    // ---- assemble scores: diagonal pos gather + tt bias + masks ----
    float pen0 = BIGINF * (1.0f - (float)amask[b * SS + j0 + colg]);
    float pen1 = BIGINF * (1.0f - (float)amask[b * SS + j0 + 16 + colg]);
    float sa0[8], sa1[8], corr[8];
#pragma unroll
    for (int vg = 0; vg < 8; ++vg) {
      int rr = vg + 8 * hi2;
      int i = i0 + rr;
      float cm0 = cls_mask[(size_t)i * SS + j0 + colg];
      float cm1 = cls_mask[(size_t)i * SS + j0 + 16 + colg];
      int t0 = token_type[((size_t)b * SS + i) * SS + j0 + colg];
      int t1 = token_type[((size_t)b * SS + i) * SS + j0 + 16 + colg];
      float pos0 = panel[rr * 48 + (colg - rr + 16)];
      float pos1 = panel[rr * 48 + (colg - rr + 32)];
      sa0[vg] = s0[vg] + (pos0 + (t0 ? ttS[vg] : ttD[vg])) * cm0 - pen0;
      sa1[vg] = s1[vg] + (pos1 + (t1 ? ttS[vg] : ttD[vg])) * cm1 - pen1;
    }

    // ---- online softmax (rows live in wave halves; xor masks stay in-half) --
#pragma unroll
    for (int vg = 0; vg < 8; ++vg) {
      float mx = fmaxf(sa0[vg], sa1[vg]);
      mx = fmaxf(mx, __shfl_xor(mx, 1));
      mx = fmaxf(mx, __shfl_xor(mx, 2));
      mx = fmaxf(mx, __shfl_xor(mx, 4));
      mx = fmaxf(mx, __shfl_xor(mx, 8));
      float mnew = fmaxf(m_r[vg], mx);
      float c = __expf(m_r[vg] - mnew);
      float p0 = __expf(sa0[vg] - mnew);
      float p1 = __expf(sa1[vg] - mnew);
      float rs = p0 + p1;
      rs += __shfl_xor(rs, 1);
      rs += __shfl_xor(rs, 2);
      rs += __shfl_xor(rs, 4);
      rs += __shfl_xor(rs, 8);
      l_r[vg] = l_r[vg] * c + rs;
      m_r[vg] = mnew;
      corr[vg] = c;
      int rr = vg + 8 * hi2;
      pstage[rr * 32 + colg] = (_Float16)p0;
      pstage[rr * 32 + 16 + colg] = (_Float16)p1;
    }
#pragma unroll
    for (int vg = 0; vg < 8; ++vg) {
      o0[vg] *= corr[vg]; o1[vg] *= corr[vg];
      o2[vg] *= corr[vg]; o3[vg] *= corr[vg];
    }

    // ---- P (C-layout) -> A-fragment via LDS (2x ds_load_b128), O += P @ V ---
    AF ap;
    {
      const _Float16* pr = &pstage[colg * 32 + (hi2 ? 8 : 0)];
      *(uint4*)&ap.u[0] = *(const uint4*)(pr);
      *(uint4*)&ap.u[4] = *(const uint4*)(pr + 16);
    }
#pragma unroll
    for (int ht = 0; ht < 4; ++ht) {
      AF bv;
      loadB16(bv, &vbb[(size_t)(ht * 16 + colg) * SS + j0], hi2);
      if (ht == 0) o0 = wmma16(ap, bv, o0);
      else if (ht == 1) o1 = wmma16(ap, bv, o1);
      else if (ht == 2) o2 = wmma16(ap, bv, o2);
      else o3 = wmma16(ap, bv, o3);
    }
  }

  // ---- normalize and write attn_vec as [b][i][n*H+h] f16 ----
#pragma unroll
  for (int vg = 0; vg < 8; ++vg) {
    int i = i0 + vg + 8 * hi2;
    float inv = 1.0f / l_r[vg];
    size_t base = ((size_t)b * SS + i) * NH + n * HH;
    av[base + 0 * 16 + colg] = (_Float16)(o0[vg] * inv);
    av[base + 1 * 16 + colg] = (_Float16)(o1[vg] * inv);
    av[base + 2 * 16 + colg] = (_Float16)(o2[vg] * inv);
    av[base + 3 * 16 + colg] = (_Float16)(o3[vg] * inv);
  }
}

// --------- Kernel 5: output projection + bias + residual --------------------
__global__ void out_gemm(const _Float16* __restrict__ av, const _Float16* __restrict__ woT,
                         const float* __restrict__ bo, const float* __restrict__ query,
                         float* xbuf) {
  int lane = threadIdx.x & 31, w = threadIdx.x >> 5;
  int it = blockIdx.x * 4 + w;
  int ct = blockIdx.y;
  int colg = lane & 15, hi2 = lane >> 4;
  int row0 = it * 16;
  int ncol = ct * 16 + colg;
  const _Float16* arow = &av[(size_t)(row0 + colg) * NH];
  const _Float16* wrow = &woT[(size_t)ncol * NH];
  v8f acc = {};
  for (int kk = 0; kk < 24; ++kk) {
    int K0 = kk * 32;
    AF a, bf;
    loadA16(a, arow + K0, hi2);
    loadB16(bf, wrow + K0, hi2);
    acc = wmma16(a, bf, acc);
  }
#pragma unroll
  for (int vg = 0; vg < 8; ++vg) {
    int mrow = row0 + vg + 8 * hi2;
    int col = ct * 16 + colg;
    xbuf[(size_t)mrow * DD + col] = acc[vg] + bo[col] + query[(size_t)mrow * DD + col];
  }
}

// --------- Kernel 6: layernorm ---------------------------------------------
__global__ void ln_kernel(const float* __restrict__ x, const float* __restrict__ g,
                          const float* __restrict__ bb, float* out) {
  int row = blockIdx.x;
  int tid = threadIdx.x;
  const float* xr = x + (size_t)row * DD;
  float v0 = xr[tid], v1 = xr[tid + 256], v2 = xr[tid + 512];
  __shared__ float red[256];
  red[tid] = v0 + v1 + v2;
  __syncthreads();
  for (int st = 128; st > 0; st >>= 1) {
    if (tid < st) red[tid] += red[tid + st];
    __syncthreads();
  }
  float mu = red[0] * (1.0f / 768.0f);
  __syncthreads();
  float d0 = v0 - mu, d1 = v1 - mu, d2 = v2 - mu;
  red[tid] = d0 * d0 + d1 * d1 + d2 * d2;
  __syncthreads();
  for (int st = 128; st > 0; st >>= 1) {
    if (tid < st) red[tid] += red[tid + st];
    __syncthreads();
  }
  float rs = rsqrtf(red[0] * (1.0f / 768.0f) + 1e-9f);
  float* orow = out + (size_t)row * DD;
  orow[tid] = d0 * rs * g[tid] + bb[tid];
  orow[tid + 256] = d1 * rs * g[tid + 256] + bb[tid + 256];
  orow[tid + 512] = d2 * rs * g[tid + 512] + bb[tid + 512];
}

extern "C" void kernel_launch(void* const* d_in, const int* in_sizes, int n_in,
                              void* d_out, int out_size, void* d_ws, size_t ws_size,
                              hipStream_t stream) {
  const float* query = (const float*)d_in[0];
  const float* key   = (const float*)d_in[1];
  const float* value = (const float*)d_in[2];
  const float* r     = (const float*)d_in[3];
  const float* cls_mask = (const float*)d_in[4];
  const int*   token_type = (const int*)d_in[5];
  const int*   amask = (const int*)d_in[6];
  const float* wq = (const float*)d_in[7];
  const float* wk = (const float*)d_in[8];
  const float* bk = (const float*)d_in[9];
  const float* wv = (const float*)d_in[10];
  const float* bv = (const float*)d_in[11];
  const float* rwb = (const float*)d_in[12];
  const float* rrb = (const float*)d_in[13];
  const float* rsb = (const float*)d_in[14];
  const float* rkern = (const float*)d_in[15];
  const float* seg = (const float*)d_in[16];
  const float* wo = (const float*)d_in[17];
  const float* bo = (const float*)d_in[18];
  const float* ln_g = (const float*)d_in[19];
  const float* ln_b = (const float*)d_in[20];

  char* ws = (char*)d_ws;
  size_t off = 0;
  auto alloc = [&](size_t bytes) {
    size_t o = off;
    off += (bytes + 255) & ~(size_t)255;
    return o;
  };
  size_t o_wqT = alloc((size_t)DD * NH * 2);
  size_t o_wkT = alloc((size_t)DD * NH * 2);
  size_t o_wvT = alloc((size_t)DD * NH * 2);
  size_t o_rkT = alloc((size_t)DD * NH * 2);
  size_t o_woT = alloc((size_t)NH * DD * 2);
  size_t o_qw = alloc((size_t)BB * NN * SS * HH * 2);
  size_t o_qr = alloc((size_t)BB * NN * SS * HH * 2);
  size_t o_qs = alloc((size_t)BB * NN * SS * HH * 2);
  size_t o_kb = alloc((size_t)BB * NN * SS * HH * 2);
  size_t o_vT = alloc((size_t)BB * NN * SS * HH * 2);
  size_t o_rh = alloc((size_t)NN * TT * HH * 2);
  size_t o_av = alloc((size_t)BB * SS * NH * 2);
  size_t o_ttS = alloc((size_t)BB * NN * SS * 4);
  size_t o_ttD = alloc((size_t)BB * NN * SS * 4);
  size_t o_x = alloc((size_t)BB * SS * DD * 4);

  _Float16* wqT = (_Float16*)(ws + o_wqT);
  _Float16* wkT = (_Float16*)(ws + o_wkT);
  _Float16* wvT = (_Float16*)(ws + o_wvT);
  _Float16* rkT = (_Float16*)(ws + o_rkT);
  _Float16* woT = (_Float16*)(ws + o_woT);
  _Float16* qw = (_Float16*)(ws + o_qw);
  _Float16* qr = (_Float16*)(ws + o_qr);
  _Float16* qs = (_Float16*)(ws + o_qs);
  _Float16* kb_ = (_Float16*)(ws + o_kb);
  _Float16* vT = (_Float16*)(ws + o_vT);
  _Float16* rh = (_Float16*)(ws + o_rh);
  _Float16* av = (_Float16*)(ws + o_av);
  float* ttS = (float*)(ws + o_ttS);
  float* ttD = (float*)(ws + o_ttD);
  float* xbuf = (float*)(ws + o_x);

  transpose_w<<<dim3((DD * NH) / 256, 5), 256, 0, stream>>>(
      wq, wk, wv, rkern, wo, wqT, wkT, wvT, rkT, woT);
  proj_gemm<<<dim3(32, 48, 4), 128, 0, stream>>>(
      query, key, value, r, wqT, wkT, wvT, rkT, bk, bv, rwb, rrb, rsb,
      qw, qr, qs, kb_, vT, rh);
  tt_kernel<<<dim3((BB * NN * SS) / 256), 256, 0, stream>>>(qs, seg, ttS, ttD);
  attn_kernel<<<dim3((BB * NN * (SS / 16)) / 4), 128, 0, stream>>>(
      qw, qr, kb_, vT, rh, ttS, ttD, cls_mask, token_type, amask, av);
  out_gemm<<<dim3(32, 48), 128, 0, stream>>>(av, woT, bo, query, xbuf);
  ln_kernel<<<dim3(BB * SS), 256, 0, stream>>>(xbuf, ln_g, ln_b, (float*)d_out);
}
